// BackboneModel_18803366821991
// MI455X (gfx1250) — compile-verified, matrix operand order checked by reading, not run on hardware
//
#include <hip/hip_runtime.h>
#include <hip/hip_bf16.h>
#include <math.h>

// ---------------------------------------------------------------------------
// Shapes (fixed by the reference)
// B=16, L=1024, H=256, N2=64, NB=6, F=4.  M = B*L = 16384.
// ---------------------------------------------------------------------------
#define Bz   16
#define Lz   1024
#define Hz   256
#define N2z  64
#define Mz   (Bz*Lz)     // 16384

typedef _Float16 v16h __attribute__((ext_vector_type(16)));
typedef _Float16 v8h  __attribute__((ext_vector_type(8)));
typedef float    v8f  __attribute__((ext_vector_type(8)));
typedef float    v4f  __attribute__((ext_vector_type(4)));
typedef unsigned int u32x4 __attribute__((ext_vector_type(4)));
typedef int          i32x8 __attribute__((ext_vector_type(8)));
typedef int          i32x4 __attribute__((ext_vector_type(4)));

static __device__ __forceinline__ v16h cat8(v8h lo, v8h hi) {
  return __builtin_shufflevector(lo, hi, 0,1,2,3,4,5,6,7,8,9,10,11,12,13,14,15);
}

// Generic pointer -> 32-bit LDS offset (AS3 pointers are LDS byte offsets).
static __device__ __forceinline__ unsigned lds_off(const void* p) {
  return (unsigned)(unsigned long long)(__attribute__((address_space(3))) const char*)p;
}

// ---------------------------------------------------------------------------
// Tensor Data Mover: 2D tile load (x = tile0 elems of 4B, y = tile1 rows,
// row stride = stride0 elems).  padbits ORed into D# group1 word0
// (pad_enable/pad_interval/pad_amount per ISA 8.4).  This toolchain's builtin
// takes 6 args: (g0 u32x4, g1 i32x8, g2 i32x4, g3 i32x4, i32x8, cpol i32);
// the trailing groups are zero (<=2D tensor -> groups 2/3 unused).
// ---------------------------------------------------------------------------
static __device__ __forceinline__ void tdm_load_2d(const float* gptr, unsigned lds,
                                                   unsigned tile0, unsigned tile1,
                                                   unsigned long long stride0,
                                                   int padbits) {
  unsigned long long ga = (unsigned long long)gptr;
  u32x4 g0;
  g0[0] = 1u;                                           // count=1, user D#
  g0[1] = lds;                                          // lds_addr (bytes)
  g0[2] = (unsigned)(ga & 0xFFFFFFFFull);               // global_addr[31:0]
  g0[3] = (unsigned)((ga >> 32) & 0x1FFFFFFull) | (2u << 30);  // [56:32], type=2
  i32x8 g1;
  g1[0] = (2 << 16) | padbits;                          // data_size=4B + pad cfg
  g1[1] = (int)((tile0 & 0xFFFFu) << 16);               // tensor_dim0[15:0]
  g1[2] = (int)((tile0 >> 16) | ((tile1 & 0xFFFFu) << 16)); // td0 hi | td1 lo
  g1[3] = (int)((tile1 >> 16) | (tile0 << 16));         // td1 hi | tile_dim0
  g1[4] = (int)(tile1 & 0xFFFFu);                       // tile_dim1 (tile_dim2=0)
  g1[5] = (int)(stride0 & 0xFFFFFFFFull);               // tensor_dim0_stride lo
  g1[6] = (int)((stride0 >> 32) & 0xFFFFull);           // stride hi (stride1=0)
  g1[7] = 0;
  i32x4 gz  = {0, 0, 0, 0};                             // groups 2/3 unused (2D)
  i32x8 gz8 = {0, 0, 0, 0, 0, 0, 0, 0};
  __builtin_amdgcn_tensor_load_to_lds(g0, g1, gz, gz, gz8, 0);
}

#define PAD_A ((1 << 20) | (4 << 22) | (7 << 25))  // pad 8 dwords every 32 dwords

// ---------------------------------------------------------------------------
// x = relu(input * W_in + b_in)   (IN_DIM==1 so it's an outer product); skip=0
// ---------------------------------------------------------------------------
__global__ __launch_bounds__(256)
void init_x(const float* __restrict__ inp, const float* __restrict__ Win,
            const float* __restrict__ bin, float* __restrict__ X,
            float* __restrict__ SKIP) {
  int idx = blockIdx.x * 256 + threadIdx.x;   // < Mz*Hz
  int h = idx & (Hz - 1);
  int m = idx >> 8;
  float v = inp[m] * Win[h] + bin[h];
  X[idx]    = v > 0.f ? v : 0.f;
  SKIP[idx] = 0.f;
}

// ---------------------------------------------------------------------------
// Time embedding stage 1: sinusoidal(128) @ W_t1 + b_t1, silu.  grid=B, blk=256
// ---------------------------------------------------------------------------
__global__ __launch_bounds__(256)
void temb1(const float* __restrict__ t, const float* __restrict__ W,
           const float* __restrict__ bias, float* __restrict__ Out) {
  int b = blockIdx.x, o = threadIdx.x;
  float tv = t[b];
  float s = bias[o];
  const float cexp = -logf(10000.f) / 63.f;   // half-1 = 63
  for (int j = 0; j < 64; j++) {
    float ang = tv * expf(j * cexp);
    s += sinf(ang) * W[j * Hz + o] + cosf(ang) * W[(64 + j) * Hz + o];
  }
  Out[b * Hz + o] = s / (1.f + expf(-s));     // silu
}

// ---------------------------------------------------------------------------
// Small dense (16 x 256) @ (256 x 256): ACT 0=none, 1=silu.  grid=B, blk=256
// ---------------------------------------------------------------------------
template <int ACT>
__global__ __launch_bounds__(256)
void dense16(const float* __restrict__ In, const float* __restrict__ W,
             const float* __restrict__ bias, float* __restrict__ Out) {
  int b = blockIdx.x, o = threadIdx.x;
  float s = bias[o];
  for (int k = 0; k < Hz; k++) s += In[b * Hz + k] * W[k * Hz + o];
  if (ACT == 1) s = s / (1.f + expf(-s));
  Out[b * Hz + o] = s;
}

// ---------------------------------------------------------------------------
// u = x + tb (broadcast over L); z = LayerNorm_H(u)*g + b.  Wave per row.
// ---------------------------------------------------------------------------
__global__ __launch_bounds__(256)
void u_ln(const float* __restrict__ X, const float* __restrict__ TB,
          const float* __restrict__ g, const float* __restrict__ beta,
          float* __restrict__ U, float* __restrict__ Z) {
  int tid = threadIdx.x, lane = tid & 31, wave = tid >> 5;
  int row = blockIdx.x * 8 + wave;            // 0..Mz-1
  int b   = row >> 10;                        // row / Lz
  const float* xr = X + (size_t)row * Hz;
  const float* tb = TB + b * Hz;
  float u[8];
  float s = 0.f;
#pragma unroll
  for (int j = 0; j < 8; j++) { int h = lane + 32 * j; u[j] = xr[h] + tb[h]; s += u[j]; }
  for (int o = 16; o; o >>= 1) s += __shfl_xor(s, o, 32);
  float mu = s * (1.f / 256.f);
  float vs = 0.f;
#pragma unroll
  for (int j = 0; j < 8; j++) { float d = u[j] - mu; vs += d * d; }
  for (int o = 16; o; o >>= 1) vs += __shfl_xor(vs, o, 32);
  float rstd = rsqrtf(vs * (1.f / 256.f) + 1e-5f);
#pragma unroll
  for (int j = 0; j < 8; j++) {
    int h = lane + 32 * j;
    U[(size_t)row * Hz + h] = u[j];
    Z[(size_t)row * Hz + h] = (u[j] - mu) * rstd * g[h] + beta[h];
  }
}

// ---------------------------------------------------------------------------
// S4D combined conv kernel generation (no FFT): wave per h, lane holds states
// n and n+32; geometric recurrence p <- p*lam over l.
//   Kc[h][1023+d] = K0[d] (d>=0),  Kc[h][1022-q] = K1[q] (q=0..1022)
// so T[m,l] = Kc[1023 + l - m]  exactly matches the reference circular conv.
// ---------------------------------------------------------------------------
__global__ __launch_bounds__(256)
void kgen(const float* __restrict__ log_dt, const float* __restrict__ A_re,
          const float* __restrict__ A_im, const float* __restrict__ C_re,
          const float* __restrict__ C_im, float* __restrict__ Kc) {
  int tid = threadIdx.x, lane = tid & 31, wave = tid >> 5;
  int h = blockIdx.x * 8 + wave;              // 0..255
  float dt = expf(log_dt[h]);
  float lamre[2], lamim[2], eore[2], eoim[2];
#pragma unroll
  for (int s = 0; s < 2; s++) {
    int n = lane + 32 * s;
    float are = A_re[h * N2z + n], aim = A_im[h * N2z + n];
    float dre = -dt * are, dim_ = dt * aim;
    float el = expf(dre);
    float lr = el * cosf(dim_), li = el * sinf(dim_);
    lamre[s] = lr; lamim[s] = li;
    float Ere = lr - 1.f, Eim = li;
    float den = are * are + aim * aim;        // |A|^2, A = -are + i*aim
    eore[s] = (-Ere * are + Eim * aim) / den; // (exp(dtA)-1)/A
    eoim[s] = (-Ere * aim - Eim * are) / den;
  }
  for (int c = 0; c < 2; c++) {
    float pre[2], pim[2];
#pragma unroll
    for (int s = 0; s < 2; s++) {
      int n = lane + 32 * s;
      float cr = C_re[(c * Hz + h) * N2z + n], ci = C_im[(c * Hz + h) * N2z + n];
      pre[s] = cr * eore[s] - ci * eoim[s];   // Bt * lam^0
      pim[s] = cr * eoim[s] + ci * eore[s];
    }
    for (int l = 0; l < Lz; l++) {
      float v = pre[0] + pre[1];
      for (int o = 16; o; o >>= 1) v += __shfl_xor(v, o, 32);
      if (lane == 0) {
        if (c == 0)          Kc[h * 2048 + 1023 + l] = 2.f * v;
        else if (l < 1023)   Kc[h * 2048 + 1022 - l] = 2.f * v;
      }
#pragma unroll
      for (int s = 0; s < 2; s++) {
        float nr = pre[s] * lamre[s] - pim[s] * lamim[s];
        float ni = pre[s] * lamim[s] + pim[s] * lamre[s];
        pre[s] = nr; pim[s] = ni;
      }
    }
  }
}

// ---------------------------------------------------------------------------
// (B,L,H) -> (B,H,L) LDS-tiled transpose
// ---------------------------------------------------------------------------
__global__ __launch_bounds__(256)
void transp(const float* __restrict__ Z, float* __restrict__ ZT) {
  __shared__ float tile[32][33];
  int b = blockIdx.z, l0 = blockIdx.y * 32, h0 = blockIdx.x * 32;
  int tx = threadIdx.x, ty0 = threadIdx.y;    // blockDim (32,8)
#pragma unroll
  for (int i = 0; i < 4; i++) {
    int ty = ty0 + 8 * i;
    tile[ty][tx] = Z[((size_t)b * Lz + l0 + ty) * Hz + h0 + tx];
  }
  __syncthreads();
#pragma unroll
  for (int i = 0; i < 4; i++) {
    int ty = ty0 + 8 * i;
    ZT[((size_t)b * Hz + h0 + ty) * Lz + l0 + tx] = tile[tx][ty];
  }
}

// ---------------------------------------------------------------------------
// S4 conv as per-h Toeplitz WMMA GEMM: Y(16 x 1024) = Zt_h(16 x 1024) @ T_h.
// One h per block; the 8KB Kc row is DMA'd into LDS by the Tensor Data Mover.
// B-fragments synthesized from LDS.  Epilogue fuses +D*z and exact gelu,
// scatters result back into (B,L,H) layout.  grid = (Hz, 8), block = 256.
// ---------------------------------------------------------------------------
__global__ __launch_bounds__(256)
void conv_wmma(const float* __restrict__ ZT, const float* __restrict__ Kc,
               const float* __restrict__ Dvec, float* __restrict__ Y) {
  __shared__ __attribute__((aligned(16))) float sK[2048];
  int tid = threadIdx.x, lane = tid & 31, wave = tid >> 5;
  int h  = blockIdx.x;                        // 0..255
  int lT = (blockIdx.y * 8 + wave) << 4;      // output-l tile base
  int laneHi = lane >> 4, m16 = lane & 15;
  int klo = laneHi * 8;
  if (tid < 32) {                             // TDM: Kc row -> LDS
    tdm_load_2d(Kc + (size_t)h * 2048, lds_off(sK), 2048u, 1u, 2048ull, 0);
    __builtin_amdgcn_s_wait_tensorcnt(0);
  }
  __syncthreads();
  const float* zrow = ZT + ((size_t)m16 * Hz + h) * Lz;  // b = m16 row
  int l = lT + m16;                           // this lane's output column
  v8f acc = {};
  for (int k0 = 0; k0 < Lz; k0 += 32) {
    const float* pa = zrow + k0 + klo;
    v4f a0 = *(const v4f*)(pa);
    v4f a1 = *(const v4f*)(pa + 4);
    v4f a2 = *(const v4f*)(pa + 16);
    v4f a3 = *(const v4f*)(pa + 20);
    v16h af;
#pragma unroll
    for (int j = 0; j < 4; j++) { af[j]      = (_Float16)a0[j]; af[4 + j]  = (_Float16)a1[j]; }
#pragma unroll
    for (int j = 0; j < 4; j++) { af[8 + j]  = (_Float16)a2[j]; af[12 + j] = (_Float16)a3[j]; }
    v16h bfv;
#pragma unroll
    for (int j = 0; j < 16; j++) {
      int kl = klo + (j < 8 ? j : j + 8);     // fragment k mapping
      int m  = k0 + kl;
      bfv[j] = (_Float16)sK[1023 + l - m];
    }
    acc = __builtin_amdgcn_wmma_f32_16x16x32_f16(false, af, false, bfv,
                                                 (short)0, acc, false, false);
  }
  float Dh = Dvec[h];
#pragma unroll
  for (int r = 0; r < 8; r++) {
    int b = laneHi * 8 + r;
    float zv = ZT[((size_t)b * Hz + h) * Lz + l];
    float v = acc[r] + Dh * zv;
    float ge = 0.5f * v * (1.f + erff(v * 0.70710678118f));  // exact gelu
    Y[((size_t)b * Lz + l) * Hz + h] = ge;
  }
}

// ---------------------------------------------------------------------------
// Main WMMA GEMM: C(16384 x 256) = A(16384 x 256) @ Bw(256 x 256).
// A tiles (128 x 32 f32) are DMA'd to LDS by the TDM, double-buffered, with
// TDM padding (8 dwords per 32) giving a 40-dword LDS row pitch.  The f16
// B panel (256 x 16, transposed) is staged once.  f32 accumulate via WMMA.
// EP: 0 = +bias +U +feat@Wf +bf -> tanh*sigmoid -> Dst (gated)
//     1 = +bias +Res -> Dst (residual x update / skip accumulate)
//     3 = relu(+bias) -> Dst (head)
// ---------------------------------------------------------------------------
template <int EP>
__global__ __launch_bounds__(256)
void gemm256(const float* __restrict__ A, const float* __restrict__ Bw,
             const float* __restrict__ bias, float* __restrict__ Dst,
             const float* __restrict__ Res, const float* __restrict__ feat,
             const float* __restrict__ Wf, const float* __restrict__ bf2) {
  __shared__ __attribute__((aligned(16))) float    sAf[2 * 128 * 40];
  __shared__ __attribute__((aligned(16))) _Float16 sB[16 * 264];
  int tid = threadIdx.x;
  int wave = tid >> 5, lane = tid & 31;
  int laneHi = lane >> 4, m16 = lane & 15;
  int rowBase = blockIdx.x * 128;
  int n0 = blockIdx.y * 16;
  // B panel (256 x 16) -> transposed f16 LDS, staged once.
#pragma unroll
  for (int t = 0; t < 16; t++) {
    int idx = tid + t * 256;
    int n = idx & 15, k = idx >> 4;
    sB[n * 264 + k] = (_Float16)Bw[(size_t)k * Hz + n0 + n];
  }
  unsigned ldsA[2] = { lds_off(&sAf[0]), lds_off(&sAf[128 * 40]) };
  const float* Abase = A + (size_t)rowBase * Hz;
  if (tid < 32)                               // TDM: first A tile into buf0
    tdm_load_2d(Abase, ldsA[0], 32u, 128u, (unsigned long long)Hz, PAD_A);
  v8f acc = {};
  for (int s = 0; s < 8; s++) {
    int k0 = s * 32;
    if (tid < 32) {
      if (s + 1 < 8) {                        // prefetch next tile (other buf)
        tdm_load_2d(Abase + (k0 + 32), ldsA[(s + 1) & 1], 32u, 128u,
                    (unsigned long long)Hz, PAD_A);
        __builtin_amdgcn_s_wait_tensorcnt(1); // tile s done (loads in order)
      } else {
        __builtin_amdgcn_s_wait_tensorcnt(0); // last tile done
      }
    }
    __syncthreads();                          // tile s visible to all waves
    int klo = laneHi * 8;
    const float* pa = &sAf[(s & 1) * (128 * 40) + (wave * 16 + m16) * 40 + klo];
    v4f a0 = *(const v4f*)(pa);
    v4f a1 = *(const v4f*)(pa + 4);
    v4f a2 = *(const v4f*)(pa + 16);
    v4f a3 = *(const v4f*)(pa + 20);
    v16h af;
#pragma unroll
    for (int j = 0; j < 4; j++) { af[j]      = (_Float16)a0[j]; af[4 + j]  = (_Float16)a1[j]; }
#pragma unroll
    for (int j = 0; j < 4; j++) { af[8 + j]  = (_Float16)a2[j]; af[12 + j] = (_Float16)a3[j]; }
    const _Float16* pb = &sB[m16 * 264 + k0 + klo];
    v16h bfr = cat8(*(const v8h*)pb, *(const v8h*)(pb + 16));
    acc = __builtin_amdgcn_wmma_f32_16x16x32_f16(false, af, false, bfr,
                                                 (short)0, acc, false, false);
    __syncthreads();                          // done reading buf before reuse
  }
  int n = n0 + m16;
  float wfv[4] = {0.f, 0.f, 0.f, 0.f};
  if (EP == 0) { wfv[0] = Wf[n]; wfv[1] = Wf[Hz + n]; wfv[2] = Wf[2 * Hz + n]; wfv[3] = Wf[3 * Hz + n]; }
#pragma unroll
  for (int r = 0; r < 8; r++) {
    int m = rowBase + wave * 16 + laneHi * 8 + r;
    float v = acc[r] + bias[n];
    if (EP == 0) {
      v4f fv = *(const v4f*)&feat[(size_t)m * 4];
      v += Res[(size_t)m * Hz + n] + bf2[n] +
           fv[0] * wfv[0] + fv[1] * wfv[1] + fv[2] * wfv[2] + fv[3] * wfv[3];
      Dst[(size_t)m * Hz + n] = tanhf(v) * (1.f / (1.f + expf(-v)));
    } else if (EP == 1) {
      Dst[(size_t)m * Hz + n] = v + Res[(size_t)m * Hz + n];
    } else {
      Dst[(size_t)m * Hz + n] = fmaxf(v, 0.f);
    }
  }
}

// ---------------------------------------------------------------------------
// out[b,l] = H1[b,l,:] . Wh2 + bh2 + input[b,l].  Wave per row.
// ---------------------------------------------------------------------------
__global__ __launch_bounds__(256)
void head2(const float* __restrict__ H1, const float* __restrict__ Wh2,
           const float* __restrict__ bh2, const float* __restrict__ inp,
           float* __restrict__ Out) {
  int tid = threadIdx.x, lane = tid & 31, wave = tid >> 5;
  int row = blockIdx.x * 8 + wave;
  float s = 0.f;
#pragma unroll
  for (int j = 0; j < 8; j++) { int h = lane + 32 * j; s += H1[(size_t)row * Hz + h] * Wh2[h]; }
  for (int o = 16; o; o >>= 1) s += __shfl_xor(s, o, 32);
  if (lane == 0) Out[row] = s + bh2[0] + inp[row];
}

// ---------------------------------------------------------------------------
extern "C" void kernel_launch(void* const* d_in, const int* in_sizes, int n_in,
                              void* d_out, int out_size, void* d_ws, size_t ws_size,
                              hipStream_t stream) {
  const float* input = (const float*)d_in[0];
  const float* t     = (const float*)d_in[1];
  const float* feat  = (const float*)d_in[2];
  const float* W_in  = (const float*)d_in[3];
  const float* b_in  = (const float*)d_in[4];
  const float* W_t1  = (const float*)d_in[5];
  const float* b_t1  = (const float*)d_in[6];
  const float* W_t2  = (const float*)d_in[7];
  const float* b_t2  = (const float*)d_in[8];
  const float* Wh1   = (const float*)d_in[9];
  const float* bh1   = (const float*)d_in[10];
  const float* Wh2   = (const float*)d_in[11];
  const float* bh2   = (const float*)d_in[12];
  const float* ln_g  = (const float*)d_in[13];
  const float* ln_b  = (const float*)d_in[14];
  const float* logdt = (const float*)d_in[15];
  const float* A_re  = (const float*)d_in[16];
  const float* A_im  = (const float*)d_in[17];
  const float* C_re  = (const float*)d_in[18];
  const float* C_im  = (const float*)d_in[19];
  const float* Dp    = (const float*)d_in[20];
  const float* Wo    = (const float*)d_in[21];
  const float* bo    = (const float*)d_in[22];
  const float* Wt    = (const float*)d_in[23];
  const float* bt    = (const float*)d_in[24];
  const float* W1    = (const float*)d_in[25];
  const float* b1    = (const float*)d_in[26];
  const float* W2    = (const float*)d_in[27];
  const float* b2    = (const float*)d_in[28];
  const float* Wf    = (const float*)d_in[29];
  const float* bf    = (const float*)d_in[30];

  float* ws = (float*)d_ws;
  const size_t NT = (size_t)Mz * Hz;          // 4,194,304 floats (16 MB)
  float* X    = ws;
  float* SKIP = ws + 1 * NT;
  float* U    = ws + 2 * NT;                  // reused as H1 after the loop
  float* ZY   = ws + 3 * NT;                  // z, then reused as y (post-gelu)
  float* ZT   = ws + 4 * NT;
  float* G    = ws + 5 * NT;
  float* KC   = ws + 6 * NT;                  // 256 * 2048
  float* TH1  = KC + (size_t)Hz * 2048;
  float* TEMB = TH1 + Bz * Hz;
  float* TB   = TEMB + Bz * Hz;
  float* OUT  = (float*)d_out;

  init_x<<<Mz, 256, 0, stream>>>(input, W_in, b_in, X, SKIP);
  temb1<<<Bz, 256, 0, stream>>>(t, W_t1, b_t1, TH1);
  dense16<1><<<Bz, 256, 0, stream>>>(TH1, W_t2, b_t2, TEMB);

  for (int i = 0; i < 6; i++) {
    dense16<0><<<Bz, 256, 0, stream>>>(TEMB, Wt + (size_t)i * Hz * Hz, bt + i * Hz, TB);
    u_ln<<<Mz / 8, 256, 0, stream>>>(X, TB, ln_g + i * Hz, ln_b + i * Hz, U, ZY);
    kgen<<<Hz / 8, 256, 0, stream>>>(logdt + i * Hz, A_re + (size_t)i * Hz * N2z,
                                     A_im + (size_t)i * Hz * N2z,
                                     C_re + (size_t)i * 2 * Hz * N2z,
                                     C_im + (size_t)i * 2 * Hz * N2z, KC);
    transp<<<dim3(Hz / 32, Lz / 32, Bz), dim3(32, 8), 0, stream>>>(ZY, ZT);
    conv_wmma<<<dim3(Hz, 8), 256, 0, stream>>>(ZT, KC, Dp + i * Hz, ZY);
    gemm256<0><<<dim3(Mz / 128, Hz / 16), 256, 0, stream>>>(
        ZY, Wo + (size_t)i * Hz * Hz, bo + i * Hz, G, U, feat,
        Wf + (size_t)i * 4 * Hz, bf + i * Hz);
    gemm256<1><<<dim3(Mz / 128, Hz / 16), 256, 0, stream>>>(
        G, W1 + (size_t)i * Hz * Hz, b1 + i * Hz, X, X, nullptr, nullptr, nullptr);
    gemm256<1><<<dim3(Mz / 128, Hz / 16), 256, 0, stream>>>(
        G, W2 + (size_t)i * Hz * Hz, b2 + i * Hz, SKIP, SKIP, nullptr, nullptr, nullptr);
  }

  gemm256<3><<<dim3(Mz / 128, Hz / 16), 256, 0, stream>>>(
      SKIP, Wh1, bh1, U, nullptr, nullptr, nullptr, nullptr);
  head2<<<Mz / 8, 256, 0, stream>>>(U, Wh2, bh2, input, OUT);
}
